// RobustFill_35278861369656
// MI455X (gfx1250) — compile-verified, matrix operand order checked by reading, not run on hardware
//
#include <hip/hip_runtime.h>
#include <hip/hip_bf16.h>
#include <math.h>

#define Bsz   512
#define TIN   256
#define TOUT  128
#define Edim  64
#define Hdim  256
#define G4H   1024
#define KENC  320           // H + E   (h | x)
#define KDEC  576           // H + H + E  (h | ctx | y)
#define SA    584           // acat row stride (bf16 elems), 584 % 64 == 8 -> conflict-free frag loads
#define KVS   264           // key_v row stride (bf16 elems), multiple of 8
#define NTHR  512           // 16 waves

typedef unsigned short ushort_t;
typedef __attribute__((ext_vector_type(16))) __bf16 v16bf;
typedef __attribute__((ext_vector_type(8)))  float  v8f;

union BFrag { v16bf v; uint4 q[2]; };

static __device__ inline ushort_t f2bf(float f) {
    unsigned u = __float_as_uint(f);
    u += 0x7FFFu + ((u >> 16) & 1u);        // round-to-nearest-even
    return (ushort_t)(u >> 16);
}
static __device__ inline float bflo(unsigned d) { return __uint_as_float(d << 16); }
static __device__ inline float bfhi(unsigned d) { return __uint_as_float(d & 0xFFFF0000u); }
static __device__ inline float sigm(float x)    { return 1.0f / (1.0f + __expf(-x)); }
// branch-free tanh on the TRANS unit: tanh(x) = 1 - 2/(exp(2x)+1)
static __device__ inline float tanh_fast(float x) {
    return 1.0f - 2.0f / (__expf(2.0f * x) + 1.0f);
}

// A fragment: 16x32 bf16 tile from LDS (row-major, stride SA). ISA layout:
// lane row = lane&15, hi = lane>>4; K chunks [hi*8 .. +8) and [16+hi*8 .. +8).
static __device__ inline v16bf load_afrag(const ushort_t* base, int row, int hi, int kb) {
    BFrag f;
    const ushort_t* p = base + row * SA + kb * 32 + hi * 8;
    f.q[0] = *(const uint4*)(p);
    f.q[1] = *(const uint4*)(p + 16);
    return f.v;
}
// B fragment from fragment-major tiled weights: each 32x16 tile = 1KB contiguous,
// stored as two 512B halves; lane L reads 16B at L*16 within each half.
// ptile = Wt + nt*(K/32)*512 + lane*8  (wave-uniform base + per-lane constant offset)
static __device__ inline v16bf load_bfrag_t(const ushort_t* ptile, int kb) {
    BFrag f;
    f.q[0] = *(const uint4*)(ptile + kb * 512);
    f.q[1] = *(const uint4*)(ptile + kb * 512 + 256);
    return f.v;
}
static __device__ inline v8f wmma_bf16(v16bf a, v16bf b, v8f c) {
    return __builtin_amdgcn_wmma_f32_16x16x32_bf16(false, a, false, b, (short)0, c, false, false);
}

// ---------------------------------------------------------------------------
// K0: pack weights/embedding to bf16 in FRAGMENT-MAJOR tiled layout.
// Logical matrices (N rows x K cols):
//   Wenc (1024 x 320): [W_hh_enc | W_ih_enc]
//   Wdec (1024 x 576): [W_hh_out | W_ctx(=W_ih_out[:,E:]) | W_in(=W_ih_out[:,:E])]
//   Watt (256 x 256)
// Tiled: tile (nt,kb) at ((nt*(K/32)+kb)*512); within tile: half(256) + lane*8 + e,
// mapping to source (n = nt*16 + (lane&15), k = kb*32 + half*16 + e + (lane>>4)*8).
// ---------------------------------------------------------------------------
static __device__ inline void pack_tile_idx(int i, int K, int* n, int* k) {
    int tile = i >> 9;
    int rem  = i & 511;
    int half = rem >> 8;
    int lane = (rem >> 3) & 31;
    int e    = rem & 7;
    int KB   = K >> 5;
    int nt   = tile / KB, kb = tile - nt * KB;
    *n = nt * 16 + (lane & 15);
    *k = kb * 32 + half * 16 + e + (lane >> 4) * 8;
}

extern "C" __global__ void rf_prepack(const float* __restrict__ Wih_enc,
                                      const float* __restrict__ Whh_enc,
                                      const float* __restrict__ Watt_f,
                                      const float* __restrict__ Wih_out,
                                      const float* __restrict__ Whh_out,
                                      const float* __restrict__ emb,
                                      ushort_t* __restrict__ Wenc, ushort_t* __restrict__ Wdec,
                                      ushort_t* __restrict__ Watt, ushort_t* __restrict__ EmbB) {
    int i = blockIdx.x * blockDim.x + threadIdx.x;
    const int NE = G4H * KENC, ND = G4H * KDEC, NA = Hdim * Hdim, NB = 129 * Edim;
    if (i < NE) {
        int n, k; pack_tile_idx(i, KENC, &n, &k);
        float v = (k < Hdim) ? Whh_enc[n * Hdim + k] : Wih_enc[n * Edim + (k - Hdim)];
        Wenc[i] = f2bf(v);
    } else if ((i -= NE) < ND) {
        int n, k; pack_tile_idx(i, KDEC, &n, &k);
        float v;
        if (k < Hdim)          v = Whh_out[n * Hdim + k];
        else if (k < 2 * Hdim) v = Wih_out[n * (Edim + Hdim) + Edim + (k - Hdim)];
        else                   v = Wih_out[n * (Edim + Hdim) + (k - 2 * Hdim)];
        Wdec[i] = f2bf(v);
    } else if ((i -= ND) < NA) {
        int n, k; pack_tile_idx(i, Hdim, &n, &k);
        Watt[i] = f2bf(Watt_f[n * Hdim + k]);
    } else if ((i -= NA) < NB) {
        EmbB[i] = f2bf(emb[i]);
    }
}

// ---------------------------------------------------------------------------
// K1: persistent-tile seq2seq. One workgroup = 16 batch rows, 16 waves.
// Wave w owns H-columns [w*16, w*16+16) of i/f/g/o -> LSTM state in registers.
// ---------------------------------------------------------------------------
extern "C" __global__ void __launch_bounds__(NTHR, 1)
rf_seq2seq(const int* __restrict__ in_tok,  const int* __restrict__ in_len,
           const int* __restrict__ out_tok, const int* __restrict__ out_len,
           const float* __restrict__ b_enc, const float* __restrict__ b_att,
           const float* __restrict__ b_out,
           const ushort_t* __restrict__ Wenc, const ushort_t* __restrict__ Wdec,
           const ushort_t* __restrict__ Watt, const ushort_t* __restrict__ EmbB,
           ushort_t* __restrict__ Hall, float* __restrict__ dout) {
    __shared__ __align__(16) ushort_t acat[16 * SA];   // [h(0:256) | ctx(256:512) | y(512:576)] bf16
    __shared__ __align__(16) ushort_t keyv[16 * KVS];  // key_v bf16
    __shared__ __align__(16) float    alg[16 * 256];   // softmax weights
    __shared__ int ilen[16], olen[16];

    const int tid  = threadIdx.x;
    const int w    = tid >> 5;
    const int lane = tid & 31;
    const int n    = lane & 15;
    const int hi   = lane >> 4;
    const int b0   = blockIdx.x * 16;

    for (int i = tid; i < 16 * SA; i += NTHR) acat[i] = 0;
    if (tid < 16) { ilen[tid] = in_len[b0 + tid]; olen[tid] = out_len[b0 + tid]; }
    __syncthreads();

    const int col = w * 16 + n;                 // owned H-column
    const float bi_e = b_enc[col],       bf_e = b_enc[256 + col],
                bg_e = b_enc[512 + col], bo_e = b_enc[768 + col];
    const float bi_d = b_out[col],       bf_d = b_out[256 + col],
                bg_d = b_out[512 + col], bo_d = b_out[768 + col];
    const float batt = b_att[col];

    // Fragment-major tile bases: gate i/f/g/o are n-tiles w, 16+w, 32+w, 48+w.
    const int KBE = KENC / 32, KBD = KDEC / 32, KBA = Hdim / 32;
    const ushort_t* tWe_i = Wenc + ((size_t)(w)      * KBE) * 512 + lane * 8;
    const ushort_t* tWe_f = Wenc + ((size_t)(16 + w) * KBE) * 512 + lane * 8;
    const ushort_t* tWe_g = Wenc + ((size_t)(32 + w) * KBE) * 512 + lane * 8;
    const ushort_t* tWe_o = Wenc + ((size_t)(48 + w) * KBE) * 512 + lane * 8;
    const ushort_t* tWd_i = Wdec + ((size_t)(w)      * KBD) * 512 + lane * 8;
    const ushort_t* tWd_f = Wdec + ((size_t)(16 + w) * KBD) * 512 + lane * 8;
    const ushort_t* tWd_g = Wdec + ((size_t)(32 + w) * KBD) * 512 + lane * 8;
    const ushort_t* tWd_o = Wdec + ((size_t)(48 + w) * KBD) * 512 + lane * 8;
    const ushort_t* tWa   = Watt + ((size_t)w * KBA) * 512 + lane * 8;

    int lenI[8], lenO[8];
#pragma unroll
    for (int r = 0; r < 8; ++r) { lenI[r] = ilen[r + 8 * hi]; lenO[r] = olen[r + 8 * hi]; }

    v8f creg, hreg;
#pragma unroll
    for (int r = 0; r < 8; ++r) { creg[r] = 0.f; hreg[r] = 0.f; }

    // ================= encoder scan =================
    for (int t = 0; t < TIN; ++t) {
        if (tid < 256) {                                   // x tile gather
            int row = tid >> 4, q = tid & 15;
            int tok = in_tok[(b0 + row) * TIN + t];
            *(uint2*)(acat + row * SA + 256 + q * 4) = *(const uint2*)(EmbB + tok * Edim + q * 4);
        }
        __syncthreads();

        v8f ai, af, ag, ao;
#pragma unroll
        for (int r = 0; r < 8; ++r) { ai[r] = 0.f; af[r] = 0.f; ag[r] = 0.f; ao[r] = 0.f; }
#pragma unroll 2
        for (int kb = 0; kb < KENC / 32; ++kb) {
            v16bf A = load_afrag(acat, n, hi, kb);
            ai = wmma_bf16(A, load_bfrag_t(tWe_i, kb), ai);
            af = wmma_bf16(A, load_bfrag_t(tWe_f, kb), af);
            ag = wmma_bf16(A, load_bfrag_t(tWe_g, kb), ag);
            ao = wmma_bf16(A, load_bfrag_t(tWe_o, kb), ao);
        }
        __syncthreads();

#pragma unroll
        for (int r = 0; r < 8; ++r) {
            int m = r + 8 * hi;
            float iv = ai[r] + bi_e, fv = af[r] + bf_e, gv = ag[r] + bg_e, ov = ao[r] + bo_e;
            float cn = sigm(fv) * creg[r] + sigm(iv) * tanh_fast(gv);
            float hn = sigm(ov) * tanh_fast(cn);
            bool  mk = t < lenI[r];
            creg[r]  = mk ? cn : creg[r];
            float hc = mk ? hn : hreg[r];
            hreg[r]  = hc;
            acat[m * SA + col] = f2bf(hc);
            Hall[((size_t)t * Bsz + (b0 + m)) * Hdim + col] = f2bf(mk ? hn : 0.0f);
        }
        __syncthreads();
    }

    // ================= decoder scan =================
    for (int t = 0; t < TOUT; ++t) {
        // key_v = h @ W_att^T + b_att  (wave w -> 16 cols)
        v8f ka;
#pragma unroll
        for (int r = 0; r < 8; ++r) ka[r] = 0.f;
#pragma unroll
        for (int kb = 0; kb < Hdim / 32; ++kb) {
            v16bf A = load_afrag(acat, n, hi, kb);
            ka = wmma_bf16(A, load_bfrag_t(tWa, kb), ka);
        }
#pragma unroll
        for (int r = 0; r < 8; ++r)
            keyv[(r + 8 * hi) * KVS + col] = f2bf(ka[r] + batt);

        if (tid < 256) {                                   // y tile gather
            int row = tid >> 4, q = tid & 15;
            int tok = out_tok[(b0 + row) * TOUT + t];
            *(uint2*)(acat + row * SA + 512 + q * 4) = *(const uint2*)(EmbB + tok * Edim + q * 4);
        }
        __syncthreads();

        // scores + softmax: wave w handles batch row w, lane covers 8 t-values
        {
            const int bg = b0 + w;
            const int L  = ilen[w];
            const uint4* kp = (const uint4*)(keyv + w * KVS);
            float s[8];
#pragma unroll
            for (int tt = 0; tt < 8; ++tt) {
                int ti = lane + tt * 32;
                const uint4* hp = (const uint4*)(Hall + ((size_t)ti * Bsz + bg) * Hdim);
                float acc = 0.f;
#pragma unroll 4
                for (int hc = 0; hc < Hdim / 8; ++hc) {
                    uint4 hv = hp[hc], kv = kp[hc];
                    acc += bflo(hv.x) * bflo(kv.x) + bfhi(hv.x) * bfhi(kv.x);
                    acc += bflo(hv.y) * bflo(kv.y) + bfhi(hv.y) * bfhi(kv.y);
                    acc += bflo(hv.z) * bflo(kv.z) + bfhi(hv.z) * bfhi(kv.z);
                    acc += bflo(hv.w) * bflo(kv.w) + bfhi(hv.w) * bfhi(kv.w);
                }
                s[tt] = (ti < L) ? acc : -__builtin_inff();
            }
            float mx = s[0];
#pragma unroll
            for (int tt = 1; tt < 8; ++tt) mx = fmaxf(mx, s[tt]);
#pragma unroll
            for (int off = 16; off > 0; off >>= 1) mx = fmaxf(mx, __shfl_xor(mx, off, 32));
            float e[8], sum = 0.f;
#pragma unroll
            for (int tt = 0; tt < 8; ++tt) { e[tt] = __expf(s[tt] - mx); sum += e[tt]; }
#pragma unroll
            for (int off = 16; off > 0; off >>= 1) sum += __shfl_xor(sum, off, 32);
            float inv = 1.0f / sum;
#pragma unroll
            for (int tt = 0; tt < 8; ++tt) alg[w * 256 + lane + tt * 32] = e[tt] * inv;
        }
        // ctx = align . in_all_h : wave w, lane covers h = lane*8..+7
        {
            const int bg = b0 + w;
            const float* ap = alg + w * 256;
            float cx[8];
#pragma unroll
            for (int j = 0; j < 8; ++j) cx[j] = 0.f;
            for (int ti = 0; ti < TIN; ++ti) {
                float a = ap[ti];
                uint4 hv = *(const uint4*)(Hall + ((size_t)ti * Bsz + bg) * Hdim + lane * 8);
                cx[0] += a * bflo(hv.x); cx[1] += a * bfhi(hv.x);
                cx[2] += a * bflo(hv.y); cx[3] += a * bfhi(hv.y);
                cx[4] += a * bflo(hv.z); cx[5] += a * bfhi(hv.z);
                cx[6] += a * bflo(hv.w); cx[7] += a * bfhi(hv.w);
            }
            uint4 pk;
            pk.x = (unsigned)f2bf(cx[0]) | ((unsigned)f2bf(cx[1]) << 16);
            pk.y = (unsigned)f2bf(cx[2]) | ((unsigned)f2bf(cx[3]) << 16);
            pk.z = (unsigned)f2bf(cx[4]) | ((unsigned)f2bf(cx[5]) << 16);
            pk.w = (unsigned)f2bf(cx[6]) | ((unsigned)f2bf(cx[7]) << 16);
            *(uint4*)(acat + w * SA + 256 + lane * 8) = pk;
        }
        __syncthreads();

        // gates = [h|ctx|y] @ Wdec^T
        v8f ai, af, ag, ao;
#pragma unroll
        for (int r = 0; r < 8; ++r) { ai[r] = 0.f; af[r] = 0.f; ag[r] = 0.f; ao[r] = 0.f; }
#pragma unroll 2
        for (int kb = 0; kb < KDEC / 32; ++kb) {
            v16bf A = load_afrag(acat, n, hi, kb);
            ai = wmma_bf16(A, load_bfrag_t(tWd_i, kb), ai);
            af = wmma_bf16(A, load_bfrag_t(tWd_f, kb), af);
            ag = wmma_bf16(A, load_bfrag_t(tWd_g, kb), ag);
            ao = wmma_bf16(A, load_bfrag_t(tWd_o, kb), ao);
        }
        __syncthreads();

#pragma unroll
        for (int r = 0; r < 8; ++r) {
            int m = r + 8 * hi;
            float iv = ai[r] + bi_d, fv = af[r] + bf_d, gv = ag[r] + bg_d, ov = ao[r] + bo_d;
            float cn = sigm(fv) * creg[r] + sigm(iv) * tanh_fast(gv);
            float hn = sigm(ov) * tanh_fast(cn);
            bool  mk = t < lenO[r];
            creg[r]  = mk ? cn : creg[r];
            float hc = mk ? hn : hreg[r];
            hreg[r]  = hc;
            acat[m * SA + col] = f2bf(hc);
            dout[((size_t)t * Bsz + (b0 + m)) * Hdim + col] = mk ? hn : 0.0f;
        }
        __syncthreads();
    }

    // final h, c
    float* hfin = dout + (size_t)TOUT * Bsz * Hdim;
    float* cfin = hfin + (size_t)Bsz * Hdim;
#pragma unroll
    for (int r = 0; r < 8; ++r) {
        int m = r + 8 * hi;
        hfin[(size_t)(b0 + m) * Hdim + col] = hreg[r];
        cfin[(size_t)(b0 + m) * Hdim + col] = creg[r];
    }
}

// ---------------------------------------------------------------------------
extern "C" void kernel_launch(void* const* d_in, const int* in_sizes, int n_in,
                              void* d_out, int out_size, void* d_ws, size_t ws_size,
                              hipStream_t stream) {
    const int*   in_tok  = (const int*)  d_in[0];
    const int*   in_len  = (const int*)  d_in[1];
    const int*   out_tok = (const int*)  d_in[2];
    const int*   out_len = (const int*)  d_in[3];
    const float* emb     = (const float*)d_in[4];
    const float* Wih_enc = (const float*)d_in[5];
    const float* Whh_enc = (const float*)d_in[6];
    const float* b_enc   = (const float*)d_in[7];
    const float* Watt_f  = (const float*)d_in[8];
    const float* b_att   = (const float*)d_in[9];
    const float* Wih_out = (const float*)d_in[10];
    const float* Whh_out = (const float*)d_in[11];
    const float* b_out   = (const float*)d_in[12];

    ushort_t* Wenc = (ushort_t*)d_ws;
    ushort_t* Wdec = Wenc + (size_t)G4H * KENC;        // 327,680 bf16 (tiled)
    ushort_t* Watt = Wdec + (size_t)G4H * KDEC;        // 589,824 bf16 (tiled)
    ushort_t* EmbB = Watt + (size_t)Hdim * Hdim;       //  65,536 bf16 (tiled)
    ushort_t* Hall = EmbB + (size_t)129 * Edim;        //   8,256 bf16 -> encoder states (67.1 MB)

    int total = G4H * KENC + G4H * KDEC + Hdim * Hdim + 129 * Edim;
    rf_prepack<<<dim3((total + 255) / 256), dim3(256), 0, stream>>>(
        Wih_enc, Whh_enc, Watt_f, Wih_out, Whh_out, emb, Wenc, Wdec, Watt, EmbB);

    rf_seq2seq<<<dim3(Bsz / 16), dim3(NTHR), 0, stream>>>(
        in_tok, in_len, out_tok, out_len, b_enc, b_att, b_out,
        Wenc, Wdec, Watt, EmbB, Hall, (float*)d_out);
}
